// GraphClassifier_61022895341869
// MI455X (gfx1250) — matrix operand order repaired and checked by reading, not compiled, not run on hardware
//
#include <hip/hip_runtime.h>

#define N_NODES 50000
#define N_EDGES 800000
#define N_GRAPHS 256
#define CH 128
#define OUT_CH 10
#define BN_EPS 1e-5f

typedef __attribute__((ext_vector_type(16))) _Float16 v16h;
typedef __attribute__((ext_vector_type(8)))  _Float16 v8h;
typedef __attribute__((ext_vector_type(4)))  _Float16 v4h;
typedef __attribute__((ext_vector_type(8)))  float    v8f;

__device__ __forceinline__ void atomAdd(float* p, float v) {
  unsafeAtomicAdd(p, v);   // -> global_atomic_add_f32 (no return)
}

__global__ void k_fill(float* p, float v, int n) {
  int i = blockIdx.x * blockDim.x + threadIdx.x;
  if (i < n) p[i] = v;
}

__global__ void k_cast_f16(const float* __restrict__ in, _Float16* __restrict__ out, int n) {
  int i = blockIdx.x * blockDim.x + threadIdx.x;
  if (i < n) out[i] = (_Float16)in[i];
}

// Swizzle W [128x128] row-major f32 into the WMMA B-fragment lane layout:
// frag[((kt*8+nt)*32+lane)*16 + i] = W[kt*32 + (lane>>4)*8 + (i&7) + (i>>3)*16][nt*16 + (lane&15)]
// (B rows striped across lanes; lane = output column, 2 K-values per VGPR)
__global__ void k_convert_w(const float* __restrict__ W, _Float16* __restrict__ frag) {
  int t = blockIdx.x * blockDim.x + threadIdx.x;   // 0..16383
  if (t >= 4 * 8 * 32 * 16) return;
  int i    = t & 15;
  int lane = (t >> 4) & 31;
  int nt   = (t >> 9) & 7;
  int kt   = t >> 12;
  int k = kt * 32 + (lane >> 4) * 8 + (i & 7) + (i >> 3) * 16;
  int n = nt * 16 + (lane & 15);
  frag[t] = (_Float16)W[k * CH + n];
}

__global__ void k_degree(const int* __restrict__ dst, float* __restrict__ deg, int nE) {
  int e = blockIdx.x * blockDim.x + threadIdx.x;
  if (e < nE) atomAdd(&deg[dst[e]], 1.0f);
}

__global__ void k_rsqrt(float* d, int n) {
  int i = blockIdx.x * blockDim.x + threadIdx.x;
  if (i < n) { float v = d[i]; d[i] = v > 0.f ? rsqrtf(v) : 0.f; }
}

// H[N,128] f16 = X[N,128] f16 @ W (pre-swizzled fragments).
// One 16-row strip per block; wave w (of 8) computes column tile w.
// K=128 -> 4 x v_wmma_f32_16x16x32_f16.
__global__ void __launch_bounds__(256)
k_gemm(const _Float16* __restrict__ X, const _Float16* __restrict__ Bfrag,
       _Float16* __restrict__ H) {
  int wave = threadIdx.x >> 5;        // column tile 0..7
  int lane = threadIdx.x & 31;
  int m  = lane & 15;
  int hi = lane >> 4;
  int row = blockIdx.x * 16 + m;      // A: lanes 0-15/16-31 both carry rows 0..15
  v8f acc = {};
  #pragma unroll
  for (int kt = 0; kt < 4; ++kt) {
    // A 16x32 f16 layout: lane<16 holds K={0..7,16..23}, lane>=16 K={8..15,24..31}
    const _Float16* ap = X + row * CH + kt * 32 + hi * 8;
    v8h alo = *(const v8h*)ap;          // 16B load
    v8h ahi = *(const v8h*)(ap + 16);   // 16B load
    v16h a;
    #pragma unroll
    for (int i = 0; i < 8; ++i) { a[i] = alo[i]; a[i + 8] = ahi[i]; }
    v16h b = *(const v16h*)(Bfrag + (((kt * 8 + wave) * 32) + lane) * 16); // 32B load
    acc = __builtin_amdgcn_wmma_f32_16x16x32_f16(false, a, false, b, (short)0,
                                                 acc, false, false);
  }
  // C/D layout: VGPR r -> (M = r + hi*8, N = lane&15)
  #pragma unroll
  for (int r = 0; r < 8; ++r) {
    int mr = blockIdx.x * 16 + r + hi * 8;
    H[mr * CH + wave * 16 + m] = (_Float16)acc[r];
  }
}

// One edge per wave32; each lane handles 4 channels (8B f16 gather, 4 f32 atomics).
// Accumulates dinv[s]*h[s] into agg[dst]; dinv[d] factor applied in finalize.
__global__ void k_edge_scatter(const _Float16* __restrict__ H,
                               const float* __restrict__ dinv,
                               const int* __restrict__ src,
                               const int* __restrict__ dst,
                               float* __restrict__ agg, int nE) {
  int gid = blockIdx.x * blockDim.x + threadIdx.x;
  int e = gid >> 5;
  if (e >= nE) return;
  int lane = gid & 31;
  int s = src[e], d = dst[e];
  float c = dinv[s];
  v4h hv = *(const v4h*)(H + s * CH + lane * 4);
  float* ap = agg + d * CH + lane * 4;
  #pragma unroll
  for (int j = 0; j < 4; ++j) atomAdd(ap + j, c * (float)hv[j]);
}

// z = dinv[d]*agg + dinv[d]^2*h (self loop) + bias; in-place into agg.
// Also produces block-partial BN sum / sumsq (one channel per thread, 64 rows).
__global__ void k_finalize(float* __restrict__ agg, const _Float16* __restrict__ H,
                           const float* __restrict__ dinv, const float* __restrict__ bias,
                           float* __restrict__ sums) {
  int c = threadIdx.x;                // 0..127
  int row0 = blockIdx.x * 64;
  float b = bias[c];
  float ls = 0.f, lq = 0.f;
  for (int r = 0; r < 64; ++r) {
    int i = row0 + r;
    if (i >= N_NODES) break;
    float di = dinv[i];
    float z = di * agg[i * CH + c] + di * di * (float)H[i * CH + c] + b;
    agg[i * CH + c] = z;
    ls += z; lq += z * z;
  }
  atomAdd(&sums[c], ls);
  atomAdd(&sums[CH + c], lq);
}

__global__ void k_bn_params(const float* __restrict__ sums, const float* __restrict__ g,
                            const float* __restrict__ be, float* __restrict__ params) {
  int c = threadIdx.x;
  float mean = sums[c] / (float)N_NODES;
  float var = sums[CH + c] / (float)N_NODES - mean * mean;
  float sc = g[c] * rsqrtf(var + BN_EPS);
  params[c] = sc;
  params[CH + c] = be[c] - mean * sc;
}

// y = relu(z*scale+shift); keep f32 in-place (readout) and f16 copy (next GEMM).
__global__ void k_apply(float* __restrict__ z, const float* __restrict__ params,
                        _Float16* __restrict__ xnext, int n) {
  int i = blockIdx.x * blockDim.x + threadIdx.x;
  if (i >= n) return;
  int c = i & (CH - 1);
  float y = fmaxf(z[i] * params[c] + params[CH + c], 0.f);
  z[i] = y;
  xnext[i] = (_Float16)y;
}

// One node per wave32, 4 channels per lane -> per-graph sums.
__global__ void k_readout(const float* __restrict__ y, const int* __restrict__ batch,
                          float* __restrict__ gsum, float* __restrict__ gcnt) {
  int gid = blockIdx.x * blockDim.x + threadIdx.x;
  int i = gid >> 5;
  if (i >= N_NODES) return;
  int lane = gid & 31;
  int b = batch[i];
  const float* yp = y + i * CH + lane * 4;
  float* gp = gsum + b * CH + lane * 4;
  #pragma unroll
  for (int j = 0; j < 4; ++j) atomAdd(gp + j, yp[j]);
  if (lane == 0) atomAdd(&gcnt[b], 1.0f);
}

// Per-graph MLP: h = relu(concat(mean,sum) @ w1 + b1); out = h @ w2 + b2.
__global__ void k_mlp(const float* __restrict__ gsum, const float* __restrict__ gcnt,
                      const float* __restrict__ w1, const float* __restrict__ b1,
                      const float* __restrict__ w2, const float* __restrict__ b2,
                      float* __restrict__ out) {
  __shared__ float a[CH];
  int g = blockIdx.x, j = threadIdx.x;
  float cnt = fmaxf(gcnt[g], 1.0f);
  float acc = b1[j];
  for (int k = 0; k < CH; ++k) {
    float s = gsum[g * CH + k];
    acc += (s / cnt) * w1[k * CH + j] + s * w1[(CH + k) * CH + j];
  }
  a[j] = fmaxf(acc, 0.f);
  __syncthreads();
  if (j < OUT_CH) {
    float o = b2[j];
    for (int k = 0; k < CH; ++k) o += a[k] * w2[k * OUT_CH + j];
    out[g * OUT_CH + j] = o;
  }
}

extern "C" void kernel_launch(void* const* d_in, const int* in_sizes, int n_in,
                              void* d_out, int out_size, void* d_ws, size_t ws_size,
                              hipStream_t stream) {
  (void)in_sizes; (void)n_in; (void)out_size; (void)ws_size;
  const float* x     = (const float*)d_in[0];
  const int*   ei    = (const int*)d_in[1];
  const int*   batch = (const int*)d_in[2];
  const float* W[3]  = {(const float*)d_in[3],  (const float*)d_in[7],  (const float*)d_in[11]};
  const float* bb[3] = {(const float*)d_in[4],  (const float*)d_in[8],  (const float*)d_in[12]};
  const float* gg[3] = {(const float*)d_in[5],  (const float*)d_in[9],  (const float*)d_in[13]};
  const float* be[3] = {(const float*)d_in[6],  (const float*)d_in[10], (const float*)d_in[14]};
  const float* w1 = (const float*)d_in[15];
  const float* b1 = (const float*)d_in[16];
  const float* w2 = (const float*)d_in[17];
  const float* b2 = (const float*)d_in[18];
  const int* src = ei;
  const int* dst = ei + N_EDGES;
  float* out = (float*)d_out;

  size_t off = 0;
  auto alloc = [&](size_t bytes) -> char* {
    char* p = (char*)d_ws + off;
    off += (bytes + 255) & ~(size_t)255;
    return p;
  };
  _Float16* xf16  = (_Float16*)alloc((size_t)N_NODES * CH * 2);   // GEMM input
  _Float16* hf16  = (_Float16*)alloc((size_t)N_NODES * CH * 2);   // GEMM output
  float*    agg   = (float*)alloc((size_t)N_NODES * CH * 4);      // agg -> z -> y
  float*    dinv  = (float*)alloc((size_t)N_NODES * 4);           // deg -> rsqrt(deg)
  _Float16* bfrag = (_Float16*)alloc((size_t)3 * 16384 * 2);      // swizzled W0..W2
  float*    sums  = (float*)alloc(256 * 4);                       // [sum(128)|sumsq(128)]
  float*    prm   = (float*)alloc(256 * 4);                       // [scale|shift]
  float*    gsum  = (float*)alloc((size_t)N_GRAPHS * CH * 4);
  float*    gcnt  = (float*)alloc(N_GRAPHS * 4);

  const int NC = N_NODES * CH;   // 6,400,000

  for (int l = 0; l < 3; ++l)
    k_convert_w<<<64, 256, 0, stream>>>(W[l], bfrag + l * 16384);
  k_cast_f16<<<(NC + 255) / 256, 256, 0, stream>>>(x, xf16, NC);

  // degrees (init 1.0 = self loop), then dinv = rsqrt(deg)
  k_fill<<<(N_NODES + 255) / 256, 256, 0, stream>>>(dinv, 1.0f, N_NODES);
  k_degree<<<(N_EDGES + 255) / 256, 256, 0, stream>>>(dst, dinv, N_EDGES);
  k_rsqrt<<<(N_NODES + 255) / 256, 256, 0, stream>>>(dinv, N_NODES);

  for (int l = 0; l < 3; ++l) {
    k_fill<<<(NC + 255) / 256, 256, 0, stream>>>(agg, 0.0f, NC);
    k_gemm<<<N_NODES / 16, 256, 0, stream>>>(xf16, bfrag + l * 16384, hf16);
    k_edge_scatter<<<(N_EDGES * 32) / 256, 256, 0, stream>>>(hf16, dinv, src, dst, agg, N_EDGES);
    k_fill<<<1, 256, 0, stream>>>(sums, 0.0f, 256);
    k_finalize<<<(N_NODES + 63) / 64, 128, 0, stream>>>(agg, hf16, dinv, bb[l], sums);
    k_bn_params<<<1, 128, 0, stream>>>(sums, gg[l], be[l], prm);
    k_apply<<<(NC + 255) / 256, 256, 0, stream>>>(agg, prm, xf16, NC);
  }

  k_fill<<<(N_GRAPHS * CH + 255) / 256, 256, 0, stream>>>(gsum, 0.0f, N_GRAPHS * CH);
  k_fill<<<1, 256, 0, stream>>>(gcnt, 0.0f, N_GRAPHS);
  k_readout<<<(N_NODES * 32 + 255) / 256, 256, 0, stream>>>(agg, batch, gsum, gcnt);
  k_mlp<<<N_GRAPHS, CH, 0, stream>>>(gsum, gcnt, w1, b1, w2, b2, out);
}
